// TimeSeriesTransformerPooled_23175643529607
// MI455X (gfx1250) — matrix operand order repaired and checked by reading, not compiled, not run on hardware
//
#include <hip/hip_runtime.h>
#include <cstdint>

// ---------------- problem constants ----------------
#define BB   32
#define SS   512
#define INP  64
#define DD   512
#define HH   8
#define LL   4
#define FFD  2048
#define HDIM 64

typedef __attribute__((ext_vector_type(16))) __bf16 bf16x16;
typedef __attribute__((ext_vector_type(8)))  float  floatx8;

struct __align__(16) U4 { unsigned int a, b, c, d; };
struct __align__(16) F4 { float x, y, z, w; };
union FragU { U4 q[2]; bf16x16 v; };

__device__ __forceinline__ unsigned short f2bf(float f) {
  unsigned int u = __float_as_uint(f);
  u += 0x7FFFu + ((u >> 16) & 1u);          // round-to-nearest-even
  return (unsigned short)(u >> 16);
}
__device__ __forceinline__ float gelu_f(float x) {
  return 0.5f * x * (1.0f + erff(x * 0.70710678118654752f));
}

// ---------------------------------------------------------------------------
// Generic batched GEMM:  C = act(A @ opB + bias)
//   BT==1: B is [N,K] row-major (C = A @ B^T)   -- weight layout
//   BT==0: B is [K,N] row-major (C = A @ B)     -- attn @ V
// Block tile 128(M) x 64(N), K-step 32, 256 threads = 8 waves,
// each wave computes a 32x32 patch as 2x2 v_wmma_f32_16x16x32_bf16 tiles.
// ---------------------------------------------------------------------------
template <int BT>
__global__ __launch_bounds__(256) void gemm_bf16_wmma(
    const float* __restrict__ A, const float* __restrict__ Bm,
    const float* __restrict__ bias, float* __restrict__ C,
    int M, int N, int K,
    long long lda, long long ldb, long long ldc,
    long long aOuter, long long aInner,
    long long bOuter, long long bInner,
    long long cOuter, long long cInner,
    int divH, int act)
{
  __shared__ __align__(16) unsigned short sA[128 * 40];  // [128][32] pad->40
  __shared__ __align__(16) unsigned short sB[64 * 40];   // [64][32]  pad->40

  const long long zo = blockIdx.y / divH, zi = blockIdx.y % divH;
  A  += aOuter * zo + aInner * zi;
  Bm += bOuter * zo + bInner * zi;
  C  += cOuter * zo + cInner * zi;

  const int tilesN = (N + 63) >> 6;
  const int m0 = (blockIdx.x / tilesN) * 128;
  const int n0 = (blockIdx.x % tilesN) * 64;

  const int t    = threadIdx.x;
  const int lane = t & 31;
  const int wave = t >> 5;
  const int wm   = (wave >> 1) * 32;   // wave M offset within block (0..96)
  const int wn   = (wave & 1) * 32;    // wave N offset within block (0/32)
  const int lrow = lane & 15;
  const int half = lane >> 4;

  floatx8 acc[2][2];
#pragma unroll
  for (int i = 0; i < 2; ++i)
#pragma unroll
    for (int j = 0; j < 2; ++j) acc[i][j] = (floatx8)(0.0f);

  for (int k0 = 0; k0 < K; k0 += 32) {
    // ---- stage A tile (128x32 f32 -> bf16 LDS) ----
    {
      const int r  = t >> 1;            // 0..127
      const int cb = (t & 1) * 16;      // 0 or 16
      long long gr = m0 + r; if (gr >= M) gr = M - 1;   // clamp (stores guarded)
      const float* ap = A + gr * lda + k0 + cb;
      unsigned short* dst = sA + r * 40 + cb;
#pragma unroll
      for (int j = 0; j < 4; ++j) {
        F4 f = *(const F4*)(ap + 4 * j);
        unsigned int lo = (unsigned)f2bf(f.x) | ((unsigned)f2bf(f.y) << 16);
        unsigned int hi = (unsigned)f2bf(f.z) | ((unsigned)f2bf(f.w) << 16);
        *(unsigned int*)(dst + 4 * j)     = lo;
        *(unsigned int*)(dst + 4 * j + 2) = hi;
      }
    }
    // ---- stage B tile ----
    if (BT) {           // B[N,K]: rows n0..n0+63, cols k0..k0+31
      const int r  = t >> 2;            // 0..63
      const int cb = (t & 3) * 8;       // 0,8,16,24
      const float* bp = Bm + (long long)(n0 + r) * ldb + k0 + cb;
      unsigned short* dst = sB + r * 40 + cb;
#pragma unroll
      for (int j = 0; j < 2; ++j) {
        F4 f = *(const F4*)(bp + 4 * j);
        unsigned int lo = (unsigned)f2bf(f.x) | ((unsigned)f2bf(f.y) << 16);
        unsigned int hi = (unsigned)f2bf(f.z) | ((unsigned)f2bf(f.w) << 16);
        *(unsigned int*)(dst + 4 * j)     = lo;
        *(unsigned int*)(dst + 4 * j + 2) = hi;
      }
    } else {            // B[K,N]: transpose into LDS as [n][k]
      const int kk = t >> 3;            // 0..31
      const int nb = (t & 7) * 8;       // 0..56
      const float* bp = Bm + (long long)(k0 + kk) * ldb + n0 + nb;
      F4 f0 = *(const F4*)(bp);
      F4 f1 = *(const F4*)(bp + 4);
      unsigned short v[8] = { f2bf(f0.x), f2bf(f0.y), f2bf(f0.z), f2bf(f0.w),
                              f2bf(f1.x), f2bf(f1.y), f2bf(f1.z), f2bf(f1.w) };
#pragma unroll
      for (int j = 0; j < 8; ++j) sB[(nb + j) * 40 + kk] = v[j];
    }
    __syncthreads();

    // ---- load fragments (ISA 16-bit A 16x32 / B 32x16 lane layouts) ----
    FragU fa[2], fb[2];
#pragma unroll
    for (int ti = 0; ti < 2; ++ti) {
      const unsigned short* p = sA + (wm + ti * 16 + lrow) * 40;
      fa[ti].q[0] = *(const U4*)(p + half * 8);        // K 0..7   / 8..15
      fa[ti].q[1] = *(const U4*)(p + 16 + half * 8);   // K 16..23 / 24..31
    }
#pragma unroll
    for (int tj = 0; tj < 2; ++tj) {
      const unsigned short* p = sB + (wn + tj * 16 + lrow) * 40 + half * 8;
      fb[tj].q[0] = *(const U4*)(p);
      fb[tj].q[1] = *(const U4*)(p + 16);
    }
#pragma unroll
    for (int ti = 0; ti < 2; ++ti)
#pragma unroll
      for (int tj = 0; tj < 2; ++tj)
        acc[ti][tj] = __builtin_amdgcn_wmma_f32_16x16x32_bf16(
            false, fa[ti].v, false, fb[tj].v, (short)0, acc[ti][tj], false, false);

    __syncthreads();
  }

  // ---- epilogue: bias + optional exact GELU, guarded stores ----
#pragma unroll
  for (int tj = 0; tj < 2; ++tj) {
    const int n = n0 + wn + tj * 16 + lrow;
    const float bv = bias ? bias[n] : 0.0f;
#pragma unroll
    for (int ti = 0; ti < 2; ++ti) {
      const int mb = m0 + wm + ti * 16 + half * 8;
#pragma unroll
      for (int r = 0; r < 8; ++r) {
        const int m = mb + r;
        if (m < M) {
          float v = acc[ti][tj][r] + bv;
          if (act) v = gelu_f(v);
          C[(long long)m * ldc + n] = v;
        }
      }
    }
  }
}

// ---------------- positional encoding add (in place) ----------------
__global__ __launch_bounds__(256) void add_pe_k(float* __restrict__ x) {
  const int s = blockIdx.x % SS;
  const long long row = (long long)blockIdx.x * DD;
  const int i = threadIdx.x;                 // pair index 0..255
  const float freq = __expf((2.0f * i) * (-9.210340371976184f / (float)DD));
  const float a = (float)s * freq;
  x[row + 2 * i]     += __sinf(a);
  x[row + 2 * i + 1] += __cosf(a);
}

// ---------------- in-place scaled softmax over rows of 512 ----------------
__global__ __launch_bounds__(256) void softmax_rows_k(float* __restrict__ d, float scale) {
  __shared__ float red[256];
  float* p = d + (long long)blockIdx.x * SS;
  const int t = threadIdx.x;
  float s0 = p[t] * scale, s1 = p[t + 256] * scale;
  red[t] = fmaxf(s0, s1); __syncthreads();
  for (int o = 128; o; o >>= 1) { if (t < o) red[t] = fmaxf(red[t], red[t + o]); __syncthreads(); }
  const float mx = red[0]; __syncthreads();
  const float e0 = __expf(s0 - mx), e1 = __expf(s1 - mx);
  red[t] = e0 + e1; __syncthreads();
  for (int o = 128; o; o >>= 1) { if (t < o) red[t] += red[t + o]; __syncthreads(); }
  const float inv = 1.0f / red[0];
  p[t] = e0 * inv; p[t + 256] = e1 * inv;
}

// ---------------- fused residual add + LayerNorm (rows of 512) ----------------
__global__ __launch_bounds__(256) void add_ln_k(
    const float* __restrict__ a, const float* __restrict__ b,
    const float* __restrict__ w, const float* __restrict__ bb,
    float* __restrict__ out) {
  __shared__ float r1[256], r2[256];
  const long long row = (long long)blockIdx.x * DD;
  const int t = threadIdx.x;
  const float v0 = a[row + t] + b[row + t];
  const float v1 = a[row + t + 256] + b[row + t + 256];
  r1[t] = v0 + v1; r2[t] = v0 * v0 + v1 * v1; __syncthreads();
  for (int o = 128; o; o >>= 1) { if (t < o) { r1[t] += r1[t + o]; r2[t] += r2[t + o]; } __syncthreads(); }
  const float mean = r1[0] * (1.0f / DD);
  const float var  = r2[0] * (1.0f / DD) - mean * mean;
  const float rs   = rsqrtf(var + 1e-5f);
  out[row + t]       = (v0 - mean) * rs * w[t]       + bb[t];
  out[row + t + 256] = (v1 - mean) * rs * w[t + 256] + bb[t + 256];
}

// ---------------- mean/max pool over S ----------------
__global__ __launch_bounds__(256) void pool_k(const float* __restrict__ x, float* __restrict__ pooled) {
  const int b = blockIdx.x;
  for (int d = threadIdx.x; d < DD; d += 256) {
    float sum = 0.0f, mx = -3.402823466e38f;
    const float* p = x + ((long long)b * SS) * DD + d;
    for (int s = 0; s < SS; ++s) { float v = p[(long long)s * DD]; sum += v; mx = fmaxf(mx, v); }
    pooled[b * (2 * DD) + d]      = sum * (1.0f / SS);
    pooled[b * (2 * DD) + DD + d] = mx;
  }
}

// ---------------- final dot head ----------------
__global__ __launch_bounds__(256) void head2_k(const float* __restrict__ h,
                                               const float* __restrict__ w,
                                               const float* __restrict__ b2,
                                               float* __restrict__ out) {
  __shared__ float red[256];
  const int b = blockIdx.x, t = threadIdx.x;
  red[t] = h[b * DD + t] * w[t] + h[b * DD + t + 256] * w[t + 256];
  __syncthreads();
  for (int o = 128; o; o >>= 1) { if (t < o) red[t] += red[t + o]; __syncthreads(); }
  if (t == 0) out[b] = red[0] + b2[0];
}

// ---------------------------------------------------------------------------
static inline dim3 ggrid(int M, int N, int batches) {
  return dim3((unsigned)(((M + 127) / 128) * ((N + 63) / 64)), (unsigned)batches, 1);
}

extern "C" void kernel_launch(void* const* d_in, const int* in_sizes, int n_in,
                              void* d_out, int out_size, void* d_ws, size_t ws_size,
                              hipStream_t stream) {
  (void)in_sizes; (void)n_in; (void)out_size; (void)ws_size;
  const float* x         = (const float*)d_in[0];
  const float* w_in      = (const float*)d_in[1];
  const float* b_in      = (const float*)d_in[2];
  const float* in_proj_w = (const float*)d_in[3];
  const float* in_proj_b = (const float*)d_in[4];
  const float* out_proj_w= (const float*)d_in[5];
  const float* out_proj_b= (const float*)d_in[6];
  const float* ln1_w     = (const float*)d_in[7];
  const float* ln1_b     = (const float*)d_in[8];
  const float* ln2_w     = (const float*)d_in[9];
  const float* ln2_b     = (const float*)d_in[10];
  const float* ff1_w     = (const float*)d_in[11];
  const float* ff1_b     = (const float*)d_in[12];
  const float* ff2_w     = (const float*)d_in[13];
  const float* ff2_b     = (const float*)d_in[14];
  const float* head1_w   = (const float*)d_in[15];
  const float* head1_b   = (const float*)d_in[16];
  const float* head2_w   = (const float*)d_in[17];
  const float* head2_b   = (const float*)d_in[18];

  float* out = (float*)d_out;

  const long long M   = (long long)BB * SS;       // 16384
  const long long ASZ = M * DD;                   // 8.4M
  float* ws    = (float*)d_ws;
  float* xbuf  = ws;                              // [M,512]
  float* qkv   = xbuf + ASZ;                      // [M,1536]
  float* ctx   = qkv  + M * (3 * DD);             // [M,512]
  float* t1    = ctx  + ASZ;                      // [M,512]
  float* hbuf  = t1   + ASZ;                      // [M,2048]
  float* pooled= hbuf + M * FFD;                  // [32,1024]
  float* headh = pooled + BB * 2 * DD;            // [32,512]

  const long long qkvLD = 3 * DD;                 // 1536
  const long long SxS   = (long long)SS * SS;

  // 1) input projection + bias
  gemm_bf16_wmma<1><<<ggrid((int)M, DD, 1), 256, 0, stream>>>(
      x, w_in, b_in, xbuf, (int)M, DD, INP,
      INP, INP, DD, 0, 0, 0, 0, 0, 0, 1, 0);
  // 2) positional encoding
  add_pe_k<<<dim3((unsigned)M), 256, 0, stream>>>(xbuf);

  for (int l = 0; l < LL; ++l) {
    float* attn_l = out + 32 + (long long)l * BB * HH * SxS;  // [B,H,S,S] slice

    // 3a) QKV projection
    gemm_bf16_wmma<1><<<ggrid((int)M, 3 * DD, 1), 256, 0, stream>>>(
        xbuf, in_proj_w + (long long)l * 3 * DD * DD, in_proj_b + (long long)l * 3 * DD,
        qkv, (int)M, 3 * DD, DD,
        DD, DD, qkvLD, 0, 0, 0, 0, 0, 0, 1, 0);

    // 3b) scores = Q @ K^T  (batched over B*H, written straight into d_out)
    gemm_bf16_wmma<1><<<ggrid(SS, SS, BB * HH), 256, 0, stream>>>(
        qkv /*Q*/, qkv + DD /*K*/, nullptr, attn_l,
        SS, SS, HDIM,
        qkvLD, qkvLD, SS,
        (long long)SS * qkvLD, HDIM,            // A: b-stride, h-stride
        (long long)SS * qkvLD, HDIM,            // B: b-stride, h-stride
        (long long)HH * SxS, SxS,               // C: b-stride, h-stride
        HH, 0);

    // 3c) in-place scaled softmax (scale = 1/sqrt(HD))
    softmax_rows_k<<<dim3((unsigned)(BB * HH * SS)), 256, 0, stream>>>(attn_l, 0.125f);

    // 3d) ctx = attn @ V   (B non-transposed)
    gemm_bf16_wmma<0><<<ggrid(SS, HDIM, BB * HH), 256, 0, stream>>>(
        attn_l, qkv + 2 * DD /*V*/, nullptr, ctx,
        SS, HDIM, SS,
        SS, qkvLD, DD,
        (long long)HH * SxS, SxS,
        (long long)SS * qkvLD, HDIM,
        (long long)SS * DD, HDIM,               // ctx head h -> cols h*64
        HH, 0);

    // 3e) attention output projection
    gemm_bf16_wmma<1><<<ggrid((int)M, DD, 1), 256, 0, stream>>>(
        ctx, out_proj_w + (long long)l * DD * DD, out_proj_b + (long long)l * DD,
        t1, (int)M, DD, DD,
        DD, DD, DD, 0, 0, 0, 0, 0, 0, 1, 0);

    // 3f) x = LN(x + attn_out)
    add_ln_k<<<dim3((unsigned)M), 256, 0, stream>>>(
        xbuf, t1, ln1_w + l * DD, ln1_b + l * DD, xbuf);

    // 3g) FF1 + GELU
    gemm_bf16_wmma<1><<<ggrid((int)M, FFD, 1), 256, 0, stream>>>(
        xbuf, ff1_w + (long long)l * FFD * DD, ff1_b + (long long)l * FFD,
        hbuf, (int)M, FFD, DD,
        DD, DD, FFD, 0, 0, 0, 0, 0, 0, 1, 1);

    // 3h) FF2
    gemm_bf16_wmma<1><<<ggrid((int)M, DD, 1), 256, 0, stream>>>(
        hbuf, ff2_w + (long long)l * DD * FFD, ff2_b + (long long)l * DD,
        t1, (int)M, DD, FFD,
        FFD, FFD, DD, 0, 0, 0, 0, 0, 0, 1, 0);

    // 3i) x = LN(x + ff)
    add_ln_k<<<dim3((unsigned)M), 256, 0, stream>>>(
        xbuf, t1, ln2_w + l * DD, ln2_b + l * DD, xbuf);
  }

  // 4) mean/max pooling
  pool_k<<<dim3(BB), 256, 0, stream>>>(xbuf, pooled);

  // 5) head1 + GELU  (M=32, clamped loads / guarded stores)
  gemm_bf16_wmma<1><<<ggrid(BB, DD, 1), 256, 0, stream>>>(
      pooled, head1_w, head1_b, headh, BB, DD, 2 * DD,
      2 * DD, 2 * DD, DD, 0, 0, 0, 0, 0, 0, 1, 1);

  // 6) head2 dot -> out[0..31]
  head2_k<<<dim3(BB), 256, 0, stream>>>(headh, head2_w, head2_b, out);
}
